// swin_attn_block_89661737271823
// MI455X (gfx1250) — compile-verified, matrix operand order checked by reading, not run on hardware
//
#include <hip/hip_runtime.h>

typedef _Float16 f16;
typedef _Float16 v16h __attribute__((ext_vector_type(16)));
typedef _Float16 v8h  __attribute__((ext_vector_type(8)));
typedef float    v8f  __attribute__((ext_vector_type(8)));

namespace {

constexpr int CP = 264;   // f16 pitch (elems) for 64x256 LDS matrices (pad 8 kills bank conflicts)
constexpr int VP = 72;    // Vt pitch: [256 ch][64 tok + pad]
constexpr int PP = 72;    // per-wave softmax-probability pitch: [64][64 + pad]

// LDS layout (bytes); later phases overlay dead earlier regions (barriers in between)
constexpr int OFF_X  = 0;                       // Xs  64x264 f16 = 33792   (phase 0-1)
constexpr int OFF_Q  = 64 * CP * 2;             // Qs  64x264 f16           (phase 1-2)
constexpr int OFF_K  = 2 * 64 * CP * 2;         // Ks  64x264 f16           (phase 1-2)
constexpr int OFF_PS = 0;                       // Ps  8 waves x 64x72 f16 = 73728 (overlays X/Q/K)
constexpr int OFF_OT = 0;                       // OT  64x264 f32 = 67584   (overlays Ps)
constexpr int OFF_V  = 3 * 64 * CP * 2;         // Vt  256x72 f16 = 36864   (phase 1-3)
constexpr int OFF_O  = OFF_V;                   // Os  64x264 f16 (overlays Vt after PV)
constexpr int OFF_B  = OFF_V + 256 * VP * 2;    // bias table 225x8 f32 = 7200
constexpr int SMEM_BYTES = OFF_B + 225 * 8 * 4; // 145440 B  (< 320KB/WGP)

__device__ inline v16h frag16(const f16* p) {
  // WMMA A/B operand: two contiguous 8xf16 chunks at k-offset 0 and +16 (per ISA VGPR layout)
  v8h lo = *(const v8h*)p;
  v8h hi = *(const v8h*)(p + 16);
  return __builtin_shufflevector(lo, hi, 0,1,2,3,4,5,6,7,8,9,10,11,12,13,14,15);
}

__device__ inline v8f wmma_f16(v16h a, v16h b, v8f c) {
  return __builtin_amdgcn_wmma_f32_16x16x32_f16(false, a, false, b, (short)0, c, false, false);
}

__global__ void cvt_weights_kernel(const float* __restrict__ qw,
                                   const float* __restrict__ pw,
                                   f16* __restrict__ q16, f16* __restrict__ p16) {
  int i = blockIdx.x * 256 + threadIdx.x;   // grid = 768 blocks -> 196608 = 768*256 exactly
  q16[i] = (f16)qw[i];
  if (i < 256 * 256) p16[i] = (f16)pw[i];
}

__global__ __launch_bounds__(256)
void swin_window_attn_kernel(const float* __restrict__ x,
                             const f16* __restrict__ wq16,
                             const float* __restrict__ qkv_b,
                             const f16* __restrict__ wp16,
                             const float* __restrict__ proj_b,
                             const float* __restrict__ bias_tab,
                             float* __restrict__ out) {
  extern __shared__ __align__(16) char smem[];
  f16*   Xs    = (f16*)(smem + OFF_X);
  f16*   Qs    = (f16*)(smem + OFF_Q);
  f16*   Ks    = (f16*)(smem + OFF_K);
  f16*   Vt    = (f16*)(smem + OFF_V);
  float* biasT = (float*)(smem + OFF_B);

  const int tid  = threadIdx.x;
  const int lane = tid & 31;
  const int wv   = tid >> 5;        // wave id == head id for attention phases
  const int hh   = lane >> 4;       // half-wave select
  const int lm   = lane & 15;
  const int kb   = hh * 8;          // per-lane k chunk base (ISA 16-bit A/B layout)

  const int bx   = blockIdx.x;      // 2048 windows
  const int b    = bx >> 6;
  const int widx = bx & 63;
  const int wy   = widx >> 3, wx = widx & 7;
  const long imgbase = (long)b * (256L * 4096L);

  // ---------------- Phase 0: stage bias table + window input (fp32 -> f16) ----------------
  for (int k2 = tid; k2 < 225 * 8; k2 += 256) biasT[k2] = bias_tab[k2];
  #pragma unroll
  for (int it = 0; it < 8; ++it) {
    int c = it * 32 + (tid >> 3);
    int i = tid & 7;
    const float* src = x + imgbase + (long)c * 4096 + (wy * 8 + i) * 64 + wx * 8;
    float4 r0 = *(const float4*)src;
    float4 r1 = *(const float4*)(src + 4);
    f16* dst = Xs + c;
    int t0 = i * 8;
    dst[(t0+0)*CP] = (f16)r0.x; dst[(t0+1)*CP] = (f16)r0.y;
    dst[(t0+2)*CP] = (f16)r0.z; dst[(t0+3)*CP] = (f16)r0.w;
    dst[(t0+4)*CP] = (f16)r1.x; dst[(t0+5)*CP] = (f16)r1.y;
    dst[(t0+6)*CP] = (f16)r1.z; dst[(t0+7)*CP] = (f16)r1.w;
  }
  __syncthreads();

  // ---------------- Phase 1: QKV GEMM  (64x768) = Xs(64x256) @ Wqkv^T ----------------
  const float scale = 0.17677669529663687f;   // 1/sqrt(32), folded into Q
  for (int nsel = 0; nsel < 6; ++nsel) {
    const int ntile = wv + nsel * 8;          // 0..47  (Q:0-15  K:16-31  V:32-47)
    const int obase = ntile * 16;
    v8f acc[4] = {};
    for (int ks = 0; ks < 8; ++ks) {
      v16h bf = frag16(wq16 + (obase + lm) * 256 + ks * 32 + kb);
      #pragma unroll
      for (int mt = 0; mt < 4; ++mt) {
        v16h af = frag16(Xs + (mt * 16 + lm) * CP + ks * 32 + kb);
        acc[mt] = wmma_f16(af, bf, acc[mt]);
      }
    }
    const float bo = qkv_b[obase + lm];
    if (ntile < 16) {                         // Q (scaled)
      int c = obase + lm;
      #pragma unroll
      for (int mt = 0; mt < 4; ++mt)
        #pragma unroll
        for (int r = 0; r < 8; ++r)
          Qs[(mt * 16 + r + 8 * hh) * CP + c] = (f16)((acc[mt][r] + bo) * scale);
    } else if (ntile < 32) {                  // K
      int c = obase - 256 + lm;
      #pragma unroll
      for (int mt = 0; mt < 4; ++mt)
        #pragma unroll
        for (int r = 0; r < 8; ++r)
          Ks[(mt * 16 + r + 8 * hh) * CP + c] = (f16)(acc[mt][r] + bo);
    } else {                                  // V, stored transposed: Vt[channel][token]
      int c = obase - 512 + lm;
      #pragma unroll
      for (int mt = 0; mt < 4; ++mt)
        #pragma unroll
        for (int r = 0; r < 8; ++r)
          Vt[c * VP + (mt * 16 + r + 8 * hh)] = (f16)(acc[mt][r] + bo);
    }
  }
  __syncthreads();

  // ---------------- Phase 2: scores (one head per wave), bias, softmax ----------------
  v8f s[4][4];
  {
    v16h aq[4];
    #pragma unroll
    for (int mt = 0; mt < 4; ++mt)
      aq[mt] = frag16(Qs + (mt * 16 + lm) * CP + wv * 32 + kb);
    #pragma unroll
    for (int nt = 0; nt < 4; ++nt) {
      v16h bk = frag16(Ks + (nt * 16 + lm) * CP + wv * 32 + kb);
      #pragma unroll
      for (int mt = 0; mt < 4; ++mt) {
        v8f z = {};
        s[mt][nt] = wmma_f16(aq[mt], bk, z);
      }
    }
  }
  // relative position bias: idx = (ti-ui+7)*15 + (tj-uj+7), table[idx][head]
  #pragma unroll
  for (int nt = 0; nt < 4; ++nt) {
    int u = nt * 16 + lm, ui = u >> 3, uj = u & 7;
    #pragma unroll
    for (int mt = 0; mt < 4; ++mt)
      #pragma unroll
      for (int r = 0; r < 8; ++r) {
        int t = mt * 16 + r + 8 * hh;
        int idx = ((t >> 3) - ui + 7) * 15 + ((t & 7) - uj + 7);
        s[mt][nt][r] += biasT[idx * 8 + wv];
      }
  }
  // softmax over u: lane%16 columns x 4 n-tiles per row; rows split by (mt, r, hh)
  #pragma unroll
  for (int mt = 0; mt < 4; ++mt)
    #pragma unroll
    for (int r = 0; r < 8; ++r) {
      float mx = fmaxf(fmaxf(s[mt][0][r], s[mt][1][r]), fmaxf(s[mt][2][r], s[mt][3][r]));
      for (int off = 1; off < 16; off <<= 1) mx = fmaxf(mx, __shfl_xor(mx, off, 32));
      float sm = 0.f;
      #pragma unroll
      for (int nt = 0; nt < 4; ++nt) {
        float e = __expf(s[mt][nt][r] - mx);
        s[mt][nt][r] = e;
        sm += e;
      }
      for (int off = 1; off < 16; off <<= 1) sm += __shfl_xor(sm, off, 32);
      float inv = 1.0f / sm;
      #pragma unroll
      for (int nt = 0; nt < 4; ++nt) s[mt][nt][r] *= inv;
    }
  __syncthreads();   // everyone finished reading Qs/Ks -> safe to overlay with Ps

  f16* PsW = (f16*)(smem + OFF_PS) + wv * (64 * PP);
  #pragma unroll
  for (int mt = 0; mt < 4; ++mt)
    #pragma unroll
    for (int nt = 0; nt < 4; ++nt)
      #pragma unroll
      for (int r = 0; r < 8; ++r)
        PsW[(mt * 16 + r + 8 * hh) * PP + nt * 16 + lm] = (f16)s[mt][nt][r];

  // ---------------- Phase 3: O = P @ V  (per head: 64x32, K=64) ----------------
  v8f ov[4][2] = {};
  #pragma unroll
  for (int ks = 0; ks < 2; ++ks) {
    v16h ap[4];
    #pragma unroll
    for (int mt = 0; mt < 4; ++mt)
      ap[mt] = frag16(PsW + (mt * 16 + lm) * PP + ks * 32 + kb);
    #pragma unroll
    for (int nt = 0; nt < 2; ++nt) {
      v16h bvf = frag16(Vt + (wv * 32 + nt * 16 + lm) * VP + ks * 32 + kb);
      #pragma unroll
      for (int mt = 0; mt < 4; ++mt) ov[mt][nt] = wmma_f16(ap[mt], bvf, ov[mt][nt]);
    }
  }
  __syncthreads();   // all Vt/Ps reads done -> overlay Vt with Os
  f16* Os = (f16*)(smem + OFF_O);
  #pragma unroll
  for (int mt = 0; mt < 4; ++mt)
    #pragma unroll
    for (int nt = 0; nt < 2; ++nt)
      #pragma unroll
      for (int r = 0; r < 8; ++r)
        Os[(mt * 16 + r + 8 * hh) * CP + wv * 32 + nt * 16 + lm] = (f16)ov[mt][nt][r];
  __syncthreads();

  // ---------------- Phase 4: projection GEMM (64x256) = Os @ Wproj^T ----------------
  float* OT = (float*)(smem + OFF_OT);   // f32 result, pitch CP floats (overlays Ps region)
  for (int nsel = 0; nsel < 2; ++nsel) {
    const int ntile = wv * 2 + nsel;     // 0..15
    const int cb = ntile * 16;
    v8f acc[4] = {};
    for (int ks = 0; ks < 8; ++ks) {
      v16h bf = frag16(wp16 + (cb + lm) * 256 + ks * 32 + kb);
      #pragma unroll
      for (int mt = 0; mt < 4; ++mt) {
        v16h af = frag16(Os + (mt * 16 + lm) * CP + ks * 32 + kb);
        acc[mt] = wmma_f16(af, bf, acc[mt]);
      }
    }
    const float pb = proj_b[cb + lm];
    #pragma unroll
    for (int mt = 0; mt < 4; ++mt)
      #pragma unroll
      for (int r = 0; r < 8; ++r)
        OT[(mt * 16 + r + 8 * hh) * CP + cb + lm] = acc[mt][r] + pb;
  }
  __syncthreads();

  // ---------------- Phase 5: window-reverse store to (B, C, H, W), 32B contiguous ----------------
  #pragma unroll
  for (int it = 0; it < 8; ++it) {
    int c = it * 32 + (tid >> 3);
    int i = tid & 7;
    float* dst = out + imgbase + (long)c * 4096 + (wy * 8 + i) * 64 + wx * 8;
    int t0 = i * 8;
    float4 v0, v1;
    v0.x = OT[(t0+0)*CP + c]; v0.y = OT[(t0+1)*CP + c];
    v0.z = OT[(t0+2)*CP + c]; v0.w = OT[(t0+3)*CP + c];
    v1.x = OT[(t0+4)*CP + c]; v1.y = OT[(t0+5)*CP + c];
    v1.z = OT[(t0+6)*CP + c]; v1.w = OT[(t0+7)*CP + c];
    *(float4*)dst       = v0;
    *((float4*)dst + 1) = v1;
  }
}

}  // namespace

extern "C" void kernel_launch(void* const* d_in, const int* in_sizes, int n_in,
                              void* d_out, int out_size, void* d_ws, size_t ws_size,
                              hipStream_t stream) {
  const float* x        = (const float*)d_in[0];
  const float* qkv_w    = (const float*)d_in[1];
  const float* qkv_b    = (const float*)d_in[2];
  const float* proj_w   = (const float*)d_in[3];
  const float* proj_b   = (const float*)d_in[4];
  const float* bias_tab = (const float*)d_in[5];

  f16* wq16 = (f16*)d_ws;                               // 768*256 f16 = 393216 B
  f16* wp16 = (f16*)((char*)d_ws + 768 * 256 * 2);      // 256*256 f16 = 131072 B

  cvt_weights_kernel<<<768, 256, 0, stream>>>(qkv_w, proj_w, wq16, wp16);
  swin_window_attn_kernel<<<2048, 256, SMEM_BYTES, stream>>>(
      x, wq16, qkv_b, wp16, proj_b, bias_tab, (float*)d_out);
}
